// DynamicStockClustering_29953101922746
// MI455X (gfx1250) — matrix operand order repaired and checked by reading, not compiled, not run on hardware
//
#include <hip/hip_runtime.h>
#include <hip/hip_bf16.h>

// ---------------------------------------------------------------------------
// Types for CDNA5 WMMA (wave32)
// ---------------------------------------------------------------------------
typedef __attribute__((ext_vector_type(16))) __bf16 v16bf;
typedef __attribute__((ext_vector_type(8)))  __bf16 v8bf;
typedef __attribute__((ext_vector_type(8)))  float  v8f;

#define NSTOCK 4096
#define HDIM   128
#define NHEADS 4
#define DH     32
#define NCLUS  4
#define NSUBI  4
#define SCALE  0.17677669529663687f   // 1/sqrt(32)
#define MASKNEG (-1.0e30f)

__device__ __forceinline__ v8f vzero8() {
  v8f z;
#pragma unroll
  for (int i = 0; i < 8; ++i) z[i] = 0.0f;
  return z;
}

__device__ __forceinline__ v16bf combine16(v8bf lo, v8bf hi) {
  v16bf a;
#pragma unroll
  for (int i = 0; i < 8; ++i) { a[i] = lo[i]; a[i + 8] = hi[i]; }
  return a;
}

// A-operand: 16x32 bf16 tile. lane ln=row; elements 0..7 = K half*8+i,
// elements 8..15 = K 16+half*8+i  (ISA 7.12.2 "16-bit A-Matrix 16x32").
__device__ __forceinline__ v16bf load_A_tile(const __bf16* __restrict__ A, int lda,
                                             int rowbase, int kb, int ln, int half) {
  const __bf16* p = A + (size_t)(rowbase + ln) * lda + kb + half * 8;
  v8bf lo = *(const v8bf*)p;
  v8bf hi = *(const v8bf*)(p + 16);
  return combine16(lo, hi);
}

// B-operand: 32x16 bf16 tile, row-major source (K x N, ld = ldb).
// lane l holds column n=l%16; elements 0..7 = K half*8+i, 8..15 = K 16+half*8+i.
__device__ __forceinline__ v16bf load_B_tile(const __bf16* __restrict__ B, int ldb,
                                             int kbase, int nbase, int lane) {
  int hf = lane >> 4, n = lane & 15;
  v16bf b;
#pragma unroll
  for (int s = 0; s < 16; ++s) {
    int k = (s < 8) ? (hf * 8 + s) : (16 + hf * 8 + (s - 8));
    b[s] = B[(size_t)(kbase + k) * ldb + nbase + n];
  }
  return b;
}

__device__ __forceinline__ v8f wmma_bf16(v16bf a, v16bf b, v8f c) {
  // D = A(16x32) * B(32x16) + C, f32 accumulate
  return __builtin_amdgcn_wmma_f32_16x16x32_bf16(false, a, false, b, (short)0, c,
                                                 false, false);
}

// ---------------------------------------------------------------------------
// Elementwise f32 -> bf16 conversion
// ---------------------------------------------------------------------------
__global__ void f2bf_kernel(const float* __restrict__ in, __bf16* __restrict__ out, int n) {
  int i = blockIdx.x * 256 + threadIdx.x;
  if (i < n) out[i] = (__bf16)in[i];
}

// ---------------------------------------------------------------------------
// Stage 1: normalize market columns; market_reps is (1,H,M): idx h*16+m
// grid 16 x 128
// ---------------------------------------------------------------------------
__global__ void market_norm_kernel(const float* __restrict__ mr, float* __restrict__ Mn) {
  __shared__ float red[128];
  int m = blockIdx.x, t = threadIdx.x;
  float v = mr[t * 16 + m];
  red[t] = v * v;
  __syncthreads();
  for (int s = 64; s > 0; s >>= 1) {
    if (t < s) red[t] += red[t + s];
    __syncthreads();
  }
  float inv = 1.0f / fmaxf(sqrtf(red[0]), 1e-12f);
  Mn[m * 128 + t] = v * inv;
}

__global__ void market_avg_kernel(const float* __restrict__ Mn, float* __restrict__ Mavg) {
  int t = threadIdx.x;
  float a = 0.0f;
  for (int m = 0; m < 16; ++m) a += Mn[m * 128 + t];
  Mavg[t] = a * (1.0f / 16.0f);
}

// ---------------------------------------------------------------------------
// Stage 2: per-stock inverse norm + sims (one wave per stock)
// grid 1024 x 128
// ---------------------------------------------------------------------------
__global__ void sims_kernel(const float* __restrict__ X, const float* __restrict__ Mavg,
                            float* __restrict__ xinv, float* __restrict__ sims,
                            float* __restrict__ sims_out) {
  int wid = threadIdx.x >> 5, lane = threadIdx.x & 31;
  int i = blockIdx.x * 4 + wid;
  float4 x4 = ((const float4*)(X + (size_t)i * 128))[lane];
  float ss = x4.x * x4.x + x4.y * x4.y + x4.z * x4.z + x4.w * x4.w;
#pragma unroll
  for (int d = 1; d < 32; d <<= 1) ss += __shfl_xor(ss, d, 32);
  float inv = 1.0f / fmaxf(sqrtf(ss), 1e-12f);
  float4 m4 = ((const float4*)Mavg)[lane];
  float dp = x4.x * m4.x + x4.y * m4.y + x4.z * m4.z + x4.w * m4.w;
#pragma unroll
  for (int d = 1; d < 32; d <<= 1) dp += __shfl_xor(dp, d, 32);
  if (lane == 0) {
    xinv[i] = inv;
    float s = dp * inv;
    sims[i] = s;
    sims_out[i] = s;
  }
}

// ---------------------------------------------------------------------------
// Stage 3: rank-by-count -> cluster index. grid 16 x 256
// ---------------------------------------------------------------------------
__global__ void rank_kernel(const float* __restrict__ sims, int* __restrict__ rank,
                            int* __restrict__ cluster, float* __restrict__ clus_out) {
  __shared__ float cs[NSTOCK];
  int i = blockIdx.x * 256 + threadIdx.x;
  for (int j = threadIdx.x; j < NSTOCK; j += 256) cs[j] = sims[j];
  __syncthreads();
  float si = cs[i];
  int r = 0;
  for (int j = 0; j < NSTOCK; ++j) {
    float sj = cs[j];
    r += (sj < si) || (sj == si && j < i);
  }
  int rc = r >> 10;
  if (rc > 3) rc = 3;
  int c = 3 - rc;
  rank[i] = r;
  cluster[i] = c;
  clus_out[i] = (float)c;
}

// ---------------------------------------------------------------------------
// Stage 4: per-cluster normalized centroid. grid 4 x 128
// ---------------------------------------------------------------------------
__global__ void centroid_kernel(const float* __restrict__ X, const int* __restrict__ cluster,
                                float* __restrict__ Cn) {
  __shared__ float red[128];
  int c = blockIdx.x, h = threadIdx.x;
  float acc = 0.0f;
  for (int i = 0; i < NSTOCK; ++i)
    if (cluster[i] == c) acc += X[(size_t)i * 128 + h];
  float cen = acc * (1.0f / 1024.0f);
  red[h] = cen * cen;
  __syncthreads();
  for (int s = 64; s > 0; s >>= 1) {
    if (h < s) red[h] += red[h + s];
    __syncthreads();
  }
  float inv = 1.0f / fmaxf(sqrtf(red[0]), 1e-12f);
  Cn[c * 128 + h] = cen * inv;
}

// ---------------------------------------------------------------------------
// Stage 5: cos-sim vs own-cluster centroid (one wave per stock). grid 1024x128
// ---------------------------------------------------------------------------
__global__ void s2_kernel(const float* __restrict__ X, const float* __restrict__ Cn,
                          const int* __restrict__ cluster, const float* __restrict__ xinv,
                          float* __restrict__ s2) {
  int wid = threadIdx.x >> 5, lane = threadIdx.x & 31;
  int i = blockIdx.x * 4 + wid;
  int cl = cluster[i];
  float4 x4 = ((const float4*)(X + (size_t)i * 128))[lane];
  float4 c4 = ((const float4*)(Cn + cl * 128))[lane];
  float dp = x4.x * c4.x + x4.y * c4.y + x4.z * c4.z + x4.w * c4.w;
#pragma unroll
  for (int d = 1; d < 32; d <<= 1) dp += __shfl_xor(dp, d, 32);
  if (lane == 0) s2[i] = dp * xinv[i];
}

// ---------------------------------------------------------------------------
// Stage 6: within-cluster rank -> sub-interval + group id. grid 16 x 256
// ---------------------------------------------------------------------------
__global__ void subrank_kernel(const float* __restrict__ s2, const int* __restrict__ cluster,
                               const int* __restrict__ rank, int* __restrict__ group,
                               int* __restrict__ interval) {
  __shared__ float s2s[NSTOCK];
  __shared__ int   cls[NSTOCK];
  int i = blockIdx.x * 256 + threadIdx.x;
  for (int j = threadIdx.x; j < NSTOCK; j += 256) {
    s2s[j] = s2[j];
    cls[j] = cluster[j];
  }
  __syncthreads();
  float si = s2s[i];
  int cl = cls[i];
  int ri = rank[i];
  int cnt = 0;
  for (int j = 0; j < NSTOCK; ++j) {
    if (cls[j] == cl) {
      float sj = s2s[j];
      if (sj < si || (sj == si && rank[j] < ri)) cnt++;
    }
  }
  int sr = cnt >> 8;
  if (sr > 3) sr = 3;
  int sub = 3 - sr;
  group[i] = cl * 4 + sub;
  interval[i] = sub;
}

// ---------------------------------------------------------------------------
// Bmask (32 x 4096 bf16): Bmask[g][j] = (group[j]==g) ? 0 : 1.
// Used as WMMA B-operand so Cmask = Amask x Bmask = -1e30 where masked.
// grid 512 x 256
// ---------------------------------------------------------------------------
__global__ void bmask_kernel(const int* __restrict__ group, __bf16* __restrict__ Bm) {
  int idx = blockIdx.x * 256 + threadIdx.x;   // 32*4096
  int g = idx >> 12, j = idx & 4095;
  Bm[idx] = (__bf16)((group[j] == g) ? 0.0f : 1.0f);
}

// ---------------------------------------------------------------------------
// Transpose K (4096x128 bf16) -> Kt (128x4096 bf16). grid 2048 x 256
// ---------------------------------------------------------------------------
__global__ void transpose_kernel(const __bf16* __restrict__ Kb, __bf16* __restrict__ Kt) {
  int t = blockIdx.x * 256 + threadIdx.x;
  int j = t >> 7, c = t & 127;
  Kt[(size_t)c * NSTOCK + j] = Kb[t];
}

// ---------------------------------------------------------------------------
// Generic wave-tile GEMM: each wave computes 16 rows x 128 cols of
//   C = A(4096 x K bf16) @ W(K x 128 bf16) + bias(128)
// MODE 0: store bf16 (scaled by oscale) to outB. MODE 1: store f32 to outF.
// MODE 2: y = C + Xres; LayerNorm(y)*g+b -> bf16 into comb at col offset.
// Block = 128 threads (4 waves), grid = 64.
// ---------------------------------------------------------------------------
template <int K, int MODE>
__global__ void gemm16_kernel(const __bf16* __restrict__ A, int lda,
                              const __bf16* __restrict__ W,
                              const float* __restrict__ bias, float oscale,
                              __bf16* __restrict__ outB, int ldoB,
                              float* __restrict__ outF, int ldoF,
                              const float* __restrict__ Xres,
                              const float* __restrict__ lng,
                              const float* __restrict__ lnb,
                              __bf16* __restrict__ comb, int combOff, int ldcomb) {
  int wid = threadIdx.x >> 5;
  int lane = threadIdx.x & 31;
  int half = lane >> 4, ln = lane & 15;
  int rowbase = (blockIdx.x * 4 + wid) * 16;
  constexpr int KT = K / 32;

  v16bf a[KT];
#pragma unroll
  for (int kk = 0; kk < KT; ++kk) a[kk] = load_A_tile(A, lda, rowbase, kk * 32, ln, half);

  if constexpr (MODE == 2) {
    v8f c[8];
#pragma unroll
    for (int nb = 0; nb < 8; ++nb) {
      float bv = bias[nb * 16 + ln];
      v8f acc;
#pragma unroll
      for (int r = 0; r < 8; ++r) acc[r] = bv;
#pragma unroll
      for (int kk = 0; kk < KT; ++kk)
        acc = wmma_bf16(a[kk], load_B_tile(W, 128, kk * 32, nb * 16, lane), acc);
      c[nb] = acc;
    }
    // residual add
#pragma unroll
    for (int nb = 0; nb < 8; ++nb)
#pragma unroll
      for (int r = 0; r < 8; ++r) {
        int row = rowbase + r + half * 8;
        c[nb][r] += Xres[(size_t)row * 128 + nb * 16 + ln];
      }
    // layer norm stats per row (reduce across the 16-lane half)
    float mu[8], rstd[8];
#pragma unroll
    for (int r = 0; r < 8; ++r) {
      float s = 0.0f, q = 0.0f;
#pragma unroll
      for (int nb = 0; nb < 8; ++nb) {
        float v = c[nb][r];
        s += v;
        q += v * v;
      }
#pragma unroll
      for (int d = 1; d < 16; d <<= 1) {
        s += __shfl_xor(s, d, 32);
        q += __shfl_xor(q, d, 32);
      }
      float m = s * (1.0f / 128.0f);
      float var = q * (1.0f / 128.0f) - m * m;
      mu[r] = m;
      rstd[r] = rsqrtf(var + 1e-12f);
    }
#pragma unroll
    for (int nb = 0; nb < 8; ++nb) {
      float g = lng[nb * 16 + ln];
      float bb = lnb[nb * 16 + ln];
#pragma unroll
      for (int r = 0; r < 8; ++r) {
        int row = rowbase + r + half * 8;
        float v = (c[nb][r] - mu[r]) * rstd[r] * g + bb;
        comb[(size_t)row * ldcomb + combOff + nb * 16 + ln] = (__bf16)v;
      }
    }
  } else {
#pragma unroll
    for (int nb = 0; nb < 8; ++nb) {
      float bv = bias[nb * 16 + ln];
      v8f acc;
#pragma unroll
      for (int r = 0; r < 8; ++r) acc[r] = bv;
#pragma unroll
      for (int kk = 0; kk < KT; ++kk)
        acc = wmma_bf16(a[kk], load_B_tile(W, 128, kk * 32, nb * 16, lane), acc);
#pragma unroll
      for (int r = 0; r < 8; ++r) {
        int row = rowbase + r + half * 8;
        int col = nb * 16 + ln;
        if constexpr (MODE == 0)
          outB[(size_t)row * ldoB + col] = (__bf16)(acc[r] * oscale);
        else
          outF[(size_t)row * ldoF + col] = acc[r];
      }
    }
  }
}

// ---------------------------------------------------------------------------
// Flash-style masked attention for one interval k. Q pre-scaled by 1/sqrt(dh).
// One wave = one (head, 16-row tile). grid 256 x 128.
//
// The additive mask is produced by the MATRIX pipe, not VALU:
//   Amask[i][g] = (interval[i]==k && group[i]==g) ? -1e30 : 0   (in registers)
//   Bmask[g][j] = (group[j]==g) ? 0 : 1                          (precomputed)
//   Cmask = Amask x Bmask  -> fed as the C accumulator of the score WMMA.
// So S = Q.K^T + Cmask with zero compare/select ops in the loops.
//
// Pass 1 keeps a per-lane running max only; pass 2 accumulates unnormalized
// O = sum p_j v_j and per-lane row sums; O /= sum(p) once at the end.
// ---------------------------------------------------------------------------
__global__ void attn_kernel(const __bf16* __restrict__ Qb,  // 4096x128 (pre-scaled)
                            const __bf16* __restrict__ Kt,  // 128x4096 (transposed)
                            const __bf16* __restrict__ Vb,  // 4096x128
                            const __bf16* __restrict__ Bm,  // 32x4096 group mask
                            const int* __restrict__ group,
                            const int* __restrict__ interval, int kidx,
                            __bf16* __restrict__ ctx) {     // 4096x128 bf16
  __shared__ __bf16 ldsP[4][16 * 32];
  int wid = threadIdx.x >> 5, lane = threadIdx.x & 31;
  int half = lane >> 4, ln = lane & 15;
  int flat = blockIdx.x * 4 + wid;       // 0..1023
  int head = flat >> 8;                  // 0..3
  int tile = flat & 255;                 // 0..255
  int rowbase = tile * 16;
  __bf16* pbuf = &ldsP[wid][0];

  // Build Amask A-operand in registers: this lane's A-row is ln.
  int grpv = group[rowbase + ln];
  int inkv = (interval[rowbase + ln] == kidx) ? 1 : 0;
  v16bf amask;
#pragma unroll
  for (int i = 0; i < 16; ++i) {
    int g = (i < 8) ? (half * 8 + i) : (16 + half * 8 + (i - 8));
    amask[i] = (inkv && g == grpv) ? (__bf16)MASKNEG : (__bf16)0.0f;
  }

  v16bf qa = load_A_tile(Qb, 128, rowbase, head * DH, ln, half);
  const __bf16* ktbase = Kt + (size_t)(head * DH) * NSTOCK;
  const __bf16* bmbase = Bm;

  // ---- pass 1: per-lane row max only ----
  float rmax[8];
#pragma unroll
  for (int r = 0; r < 8; ++r) rmax[r] = -1e30f;

  for (int jb = 0; jb < NSTOCK; jb += 32) {
#pragma unroll
    for (int t = 0; t < 2; ++t) {
      v16bf bm = load_B_tile(bmbase, NSTOCK, 0, jb + t * 16, lane);
      v8f cm = wmma_bf16(amask, bm, vzero8());
      v16bf b = load_B_tile(ktbase, NSTOCK, 0, jb + t * 16, lane);
      v8f s = wmma_bf16(qa, b, cm);
#pragma unroll
      for (int r = 0; r < 8; ++r) rmax[r] = fmaxf(rmax[r], s[r]);
    }
  }
  // one cross-lane max tree per row (within the 16-lane half)
#pragma unroll
  for (int r = 0; r < 8; ++r)
#pragma unroll
    for (int d = 1; d < 16; d <<= 1)
      rmax[r] = fmaxf(rmax[r], __shfl_xor(rmax[r], d, 32));

  // ---- pass 2: unnormalized O and per-lane row sums ----
  float rsum[8];
#pragma unroll
  for (int r = 0; r < 8; ++r) rsum[r] = 0.0f;
  v8f o0 = vzero8(), o1 = vzero8();

  for (int jb = 0; jb < NSTOCK; jb += 32) {
#pragma unroll
    for (int t = 0; t < 2; ++t) {
      v16bf bm = load_B_tile(bmbase, NSTOCK, 0, jb + t * 16, lane);
      v8f cm = wmma_bf16(amask, bm, vzero8());
      v16bf b = load_B_tile(ktbase, NSTOCK, 0, jb + t * 16, lane);
      v8f s = wmma_bf16(qa, b, cm);
#pragma unroll
      for (int r = 0; r < 8; ++r) {
        float p = __expf(s[r] - rmax[r]);   // masked -> exp(-huge) = 0
        rsum[r] += p;
        int row = r + half * 8;
        pbuf[row * 32 + t * 16 + ln] = (__bf16)p;
      }
    }
    asm volatile("s_wait_dscnt 0" ::: "memory");
    // read P back in A-operand layout (local 16x32 tile)
    const __bf16* pp = pbuf + ln * 32 + half * 8;
    v8bf lo = *(const v8bf*)pp;
    v8bf hi = *(const v8bf*)(pp + 16);
    v16bf pa = combine16(lo, hi);
    v16bf bv0 = load_B_tile(Vb, 128, jb, head * DH, lane);
    v16bf bv1 = load_B_tile(Vb, 128, jb, head * DH + 16, lane);
    o0 = wmma_bf16(pa, bv0, o0);
    o1 = wmma_bf16(pa, bv1, o1);
    asm volatile("s_wait_dscnt 0" ::: "memory");
  }

  // one cross-lane sum tree per row, then normalize and store
#pragma unroll
  for (int r = 0; r < 8; ++r) {
#pragma unroll
    for (int d = 1; d < 16; d <<= 1) rsum[r] += __shfl_xor(rsum[r], d, 32);
    float inv = 1.0f / rsum[r];
    int row = rowbase + r + half * 8;
    ctx[(size_t)row * 128 + head * DH + ln] = (__bf16)(o0[r] * inv);
    ctx[(size_t)row * 128 + head * DH + 16 + ln] = (__bf16)(o1[r] * inv);
  }
}

// ---------------------------------------------------------------------------
// Host launch
// ---------------------------------------------------------------------------
extern "C" void kernel_launch(void* const* d_in, const int* in_sizes, int n_in,
                              void* d_out, int out_size, void* d_ws, size_t ws_size,
                              hipStream_t stream) {
  (void)in_sizes; (void)n_in; (void)out_size; (void)ws_size;
  const float* X   = (const float*)d_in[0];
  const float* MR  = (const float*)d_in[1];
  const float* Wq  = (const float*)d_in[2];
  const float* bq  = (const float*)d_in[3];
  const float* Wk  = (const float*)d_in[4];
  const float* bk  = (const float*)d_in[5];
  const float* Wv  = (const float*)d_in[6];
  const float* bv  = (const float*)d_in[7];
  const float* Wo  = (const float*)d_in[8];
  const float* bo  = (const float*)d_in[9];
  const float* lng = (const float*)d_in[10];
  const float* lnb = (const float*)d_in[11];
  const float* Wc  = (const float*)d_in[12];
  const float* bc  = (const float*)d_in[13];

  float* out       = (float*)d_out;
  float* out_reps  = out;                          // 4096*128
  float* out_clus  = out + NSTOCK * HDIM;          // 4096
  float* out_sims  = out + NSTOCK * HDIM + NSTOCK; // 4096

  // workspace carve-up (256B aligned)
  char* w = (char*)d_ws;
  auto alloc = [&](size_t bytes) -> char* {
    char* p = w;
    w += (bytes + 255) & ~(size_t)255;
    return p;
  };
  float* Mn     = (float*)alloc(16 * 128 * 4);
  float* Mavg   = (float*)alloc(128 * 4);
  float* xinv   = (float*)alloc(NSTOCK * 4);
  float* sims   = (float*)alloc(NSTOCK * 4);
  int*   rank   = (int*)alloc(NSTOCK * 4);
  int*   clus   = (int*)alloc(NSTOCK * 4);
  float* s2     = (float*)alloc(NSTOCK * 4);
  int*   group  = (int*)alloc(NSTOCK * 4);
  int*   intv   = (int*)alloc(NSTOCK * 4);
  float* Cn     = (float*)alloc(NCLUS * 128 * 4);
  __bf16* Xbf   = (__bf16*)alloc((size_t)NSTOCK * HDIM * 2);
  __bf16* Wqbf  = (__bf16*)alloc((size_t)4 * 128 * 128 * 2);
  __bf16* Wkbf  = (__bf16*)alloc((size_t)4 * 128 * 128 * 2);
  __bf16* Wvbf  = (__bf16*)alloc((size_t)4 * 128 * 128 * 2);
  __bf16* Wobf  = (__bf16*)alloc((size_t)4 * 128 * 128 * 2);
  __bf16* Wcbf  = (__bf16*)alloc((size_t)512 * 128 * 2);
  __bf16* Bmask = (__bf16*)alloc((size_t)32 * NSTOCK * 2);
  __bf16* Qbf   = (__bf16*)alloc((size_t)NSTOCK * HDIM * 2);
  __bf16* Kbf   = (__bf16*)alloc((size_t)NSTOCK * HDIM * 2);
  __bf16* Vbf   = (__bf16*)alloc((size_t)NSTOCK * HDIM * 2);
  __bf16* Ktbf  = (__bf16*)alloc((size_t)NSTOCK * HDIM * 2);
  __bf16* Ctxbf = (__bf16*)alloc((size_t)NSTOCK * HDIM * 2);
  __bf16* Comb  = (__bf16*)alloc((size_t)NSTOCK * 512 * 2);

  // bf16 conversions
  f2bf_kernel<<<(NSTOCK * HDIM + 255) / 256, 256, 0, stream>>>(X, Xbf, NSTOCK * HDIM);
  f2bf_kernel<<<(4 * 128 * 128 + 255) / 256, 256, 0, stream>>>(Wq, Wqbf, 4 * 128 * 128);
  f2bf_kernel<<<(4 * 128 * 128 + 255) / 256, 256, 0, stream>>>(Wk, Wkbf, 4 * 128 * 128);
  f2bf_kernel<<<(4 * 128 * 128 + 255) / 256, 256, 0, stream>>>(Wv, Wvbf, 4 * 128 * 128);
  f2bf_kernel<<<(4 * 128 * 128 + 255) / 256, 256, 0, stream>>>(Wo, Wobf, 4 * 128 * 128);
  f2bf_kernel<<<(512 * 128 + 255) / 256, 256, 0, stream>>>(Wc, Wcbf, 512 * 128);

  // clustering pipeline
  market_norm_kernel<<<16, 128, 0, stream>>>(MR, Mn);
  market_avg_kernel<<<1, 128, 0, stream>>>(Mn, Mavg);
  sims_kernel<<<NSTOCK / 4, 128, 0, stream>>>(X, Mavg, xinv, sims, out_sims);
  rank_kernel<<<NSTOCK / 256, 256, 0, stream>>>(sims, rank, clus, out_clus);
  centroid_kernel<<<NCLUS, 128, 0, stream>>>(X, clus, Cn);
  s2_kernel<<<NSTOCK / 4, 128, 0, stream>>>(X, Cn, clus, xinv, s2);
  subrank_kernel<<<NSTOCK / 256, 256, 0, stream>>>(s2, clus, rank, group, intv);
  bmask_kernel<<<(32 * NSTOCK) / 256, 256, 0, stream>>>(group, Bmask);

  // per-interval attention blocks (buffers reused; stream order serializes)
  const int GEMM_GRID = NSTOCK / (4 * 16);  // 64
  for (int k = 0; k < NSUBI; ++k) {
    const __bf16* wq = Wqbf + (size_t)k * 128 * 128;
    const __bf16* wk = Wkbf + (size_t)k * 128 * 128;
    const __bf16* wv = Wvbf + (size_t)k * 128 * 128;
    const __bf16* wo = Wobf + (size_t)k * 128 * 128;
    // Q pre-scaled by 1/sqrt(dh): scores = (Q*scale) . K
    gemm16_kernel<128, 0><<<GEMM_GRID, 128, 0, stream>>>(
        Xbf, 128, wq, bq + k * 128, SCALE, Qbf, 128, nullptr, 0, nullptr, nullptr,
        nullptr, nullptr, 0, 0);
    gemm16_kernel<128, 0><<<GEMM_GRID, 128, 0, stream>>>(
        Xbf, 128, wk, bk + k * 128, 1.0f, Kbf, 128, nullptr, 0, nullptr, nullptr,
        nullptr, nullptr, 0, 0);
    gemm16_kernel<128, 0><<<GEMM_GRID, 128, 0, stream>>>(
        Xbf, 128, wv, bv + k * 128, 1.0f, Vbf, 128, nullptr, 0, nullptr, nullptr,
        nullptr, nullptr, 0, 0);
    transpose_kernel<<<(NSTOCK * HDIM) / 256, 256, 0, stream>>>(Kbf, Ktbf);
    attn_kernel<<<256, 128, 0, stream>>>(Qbf, Ktbf, Vbf, Bmask, group, intv, k, Ctxbf);
    gemm16_kernel<128, 2><<<GEMM_GRID, 128, 0, stream>>>(
        Ctxbf, 128, wo, bo + k * 128, 1.0f, nullptr, 0, nullptr, 0, X, lng + k * 128,
        lnb + k * 128, Comb, k * 128, 512);
  }

  // final projection: out = Comb(4096x512) @ Wc(512x128) + bc
  gemm16_kernel<512, 1><<<GEMM_GRID, 128, 0, stream>>>(
      Comb, 512, Wcbf, bc, 1.0f, nullptr, 0, out_reps, 128, nullptr, nullptr,
      nullptr, nullptr, 0, 0);
}
